// AblationBaseNetwork_51230369907287
// MI455X (gfx1250) — matrix operand order repaired and checked by reading, not compiled, not run on hardware
//
#include <hip/hip_runtime.h>
#include <hip/hip_bf16.h>

// ---------------------------------------------------------------------------
// GNN forward (2 MLP encoders + 2 SAGEConv layers) for MI455X / gfx1250.
//  - Dense layers: v_wmma_f32_16x16x32_bf16, wave = 16x64 tile (4 accs,
//    A-fragment reused across 4 back-to-back WMMAs).
//  - Neighbor mean: CSR-by-destination built once per launch (no fp atomics),
//    wave-per-node gather-reduce fused with mean + bf16 convert.
//    307M global fp32 atomic RMWs (the old scatter) -> 0; gather traffic
//    (~0.6GB/layer) stays in the 192MB L2.
// ---------------------------------------------------------------------------

typedef __attribute__((ext_vector_type(16))) __bf16 v16bf;
typedef __attribute__((ext_vector_type(8)))  __bf16 v8bf;
typedef __attribute__((ext_vector_type(8)))  float  v8f;

#define N_TASKS 100000
#define N_VMS   2000
#define N_NODES 102000
#define HID     256
#define EMB     128
#define E_TV    800000
#define E_DEP   400000
#define E_ALL   (E_TV + E_DEP)

// ------------------------- small utility kernels ---------------------------

__global__ __launch_bounds__(256)
void zero_int_kernel(int* __restrict__ p, int n) {
    int i = blockIdx.x * 256 + threadIdx.x;
    if (i < n) p[i] = 0;
}

__global__ __launch_bounds__(256)
void copy_int_kernel(const int* __restrict__ a, int* __restrict__ b, int n) {
    int i = blockIdx.x * 256 + threadIdx.x;
    if (i < n) b[i] = a[i];
}

__global__ __launch_bounds__(256)
void maxc_kernel(const float* __restrict__ cores, int n, float* __restrict__ out) {
    __shared__ float s[256];
    float m = 1.0f;
    for (int i = threadIdx.x; i < n; i += 256) m = fmaxf(m, cores[i]);
    s[threadIdx.x] = m;
    __syncthreads();
    for (int k = 128; k > 0; k >>= 1) {
        if (threadIdx.x < k) s[threadIdx.x] = fmaxf(s[threadIdx.x], s[threadIdx.x + k]);
        __syncthreads();
    }
    if (threadIdx.x == 0) out[0] = s[0];
}

// Build task features [N_TASKS, 32] bf16 (6 real cols, zero-padded K to 32).
__global__ __launch_bounds__(256)
void task_feat_kernel(const float* __restrict__ ss, const float* __restrict__ sr,
                      const float* __restrict__ len, const float* __restrict__ ct,
                      const float* __restrict__ mem, const float* __restrict__ cpu,
                      const float* __restrict__ maxcP, __bf16* __restrict__ out, int n) {
    int i = blockIdx.x * 256 + threadIdx.x;
    if (i >= n) return;
    float inv = 1.0f / maxcP[0];
    float f[6];
    f[0] = ss[i]; f[1] = sr[i]; f[2] = len[i]; f[3] = ct[i];
    f[4] = mem[i] * (1.0f / 1000.0f);
    f[5] = cpu[i] * inv;
    __bf16* o = out + (size_t)i * 32;
#pragma unroll
    for (int j = 0; j < 6; ++j) o[j] = (__bf16)f[j];
#pragma unroll
    for (int j = 6; j < 32; ++j) o[j] = (__bf16)0.0f;
}

// Build VM features [N_VMS, 32] bf16 (10 real cols, zero-padded K to 32).
__global__ __launch_bounds__(256)
void vm_feat_kernel(const float* __restrict__ ct, const float* __restrict__ speed,
                    const float* __restrict__ energy, const float* __restrict__ memmb,
                    const float* __restrict__ availmem, const float* __restrict__ usedmemf,
                    const float* __restrict__ activecnt, const float* __restrict__ cores,
                    const float* __restrict__ availcores, const float* __restrict__ usedcpuf,
                    const float* __restrict__ maxcP, __bf16* __restrict__ out, int n) {
    int i = blockIdx.x * 256 + threadIdx.x;
    if (i >= n) return;
    float inv = 1.0f / maxcP[0];
    float f[10];
    f[0] = ct[i];
    f[1] = 1.0f / (speed[i] + 1e-08f);
    f[2] = energy[i];
    f[3] = memmb[i] * (1.0f / 1000.0f);
    f[4] = availmem[i] * (1.0f / 1000.0f);
    f[5] = usedmemf[i];
    f[6] = activecnt[i];
    f[7] = cores[i] * inv;
    f[8] = availcores[i] * inv;
    f[9] = usedcpuf[i];
    __bf16* o = out + (size_t)i * 32;
#pragma unroll
    for (int j = 0; j < 10; ++j) o[j] = (__bf16)f[j];
#pragma unroll
    for (int j = 10; j < 32; ++j) o[j] = (__bf16)0.0f;
}

// Swizzle fp32 weights [Korig, N] into the WMMA B-operand layout (bf16):
// out[(((nt*KT + kt)*32 + lane)*16 + j)] = W[kt*32 + (j<8 ? (lane/16)*8 + j : 16 + (lane/16)*8 + j-8), nt*16 + lane%16]
__global__ __launch_bounds__(256)
void prep_w_kernel(const float* __restrict__ W, int Korig, int Kpad, int N,
                   __bf16* __restrict__ out) {
    int idx = blockIdx.x * 256 + threadIdx.x;
    int KT = Kpad >> 5;
    int total = KT * (N >> 4) * 32;
    if (idx >= total) return;
    int lane = idx & 31;
    int tile = idx >> 5;
    int kt = tile % KT;
    int nt = tile / KT;
    int n  = nt * 16 + (lane & 15);
    int g  = lane >> 4;
    __bf16* o = out + (size_t)idx * 16;
#pragma unroll
    for (int j = 0; j < 16; ++j) {
        int k = kt * 32 + (j < 8 ? g * 8 + j : 16 + g * 8 + (j - 8));
        float v = (k < Korig) ? W[(size_t)k * N + n] : 0.0f;
        o[j] = (__bf16)v;
    }
}

// ---------------------- WMMA GEMM (wave = one 16x64 tile) ------------------
// out[M,N] = act( A1[M,K1]@W1 (+ A2[M,K2]@W2) + bias[N] (+ resid[M,N]) )
// NSUB accumulators per wave: A fragment loaded once per k-step and reused by
// NSUB independent back-to-back WMMAs (no D->A/B hazard chains).
template <int NSUB>
__global__ __launch_bounds__(32)
void gemm_wmma_kernel(const __bf16* __restrict__ A1, const __bf16* __restrict__ W1, int K1,
                      const __bf16* __restrict__ A2, const __bf16* __restrict__ W2, int K2,
                      const float* __restrict__ bias, const float* __restrict__ resid,
                      float* __restrict__ outF, __bf16* __restrict__ outB,
                      int N, int relu) {
    const int mt   = blockIdx.x;
    const int ntb  = blockIdx.y * NSUB;       // first 16-col tile of this wave
    const int lane = threadIdx.x;             // 0..31
    const int g    = lane >> 4;
    const int col  = lane & 15;

    v8f acc[NSUB];
#pragma unroll
    for (int ns = 0; ns < NSUB; ++ns) {
        float bv = bias[(ntb + ns) * 16 + col];
#pragma unroll
        for (int r = 0; r < 8; ++r) acc[ns][r] = bv;
    }

    // --- operand pair 1 ---
    {
        const int KT = K1 >> 5;
        const __bf16* arow = A1 + (size_t)(mt * 16 + col) * K1;
        for (int kt = 0; kt < KT; ++kt) {
            union { v16bf v; v8bf h[2]; } a;
            a.h[0] = *(const v8bf*)(arow + kt * 32 + g * 8);
            a.h[1] = *(const v8bf*)(arow + kt * 32 + 16 + g * 8);
#pragma unroll
            for (int ns = 0; ns < NSUB; ++ns) {
                const __bf16* bp =
                    W1 + ((size_t)((ntb + ns) * KT + kt) * 32 + lane) * 16;
                v16bf b = *(const v16bf*)bp;
                acc[ns] = __builtin_amdgcn_wmma_f32_16x16x32_bf16(
                    false, a.v, false, b, (short)0, acc[ns], false, false);
            }
        }
    }
    // --- operand pair 2 (SAGE: mean@wl + h@wr) ---
    if (A2 != nullptr) {
        const int KT = K2 >> 5;
        const __bf16* arow = A2 + (size_t)(mt * 16 + col) * K2;
        for (int kt = 0; kt < KT; ++kt) {
            union { v16bf v; v8bf h[2]; } a;
            a.h[0] = *(const v8bf*)(arow + kt * 32 + g * 8);
            a.h[1] = *(const v8bf*)(arow + kt * 32 + 16 + g * 8);
#pragma unroll
            for (int ns = 0; ns < NSUB; ++ns) {
                const __bf16* bp =
                    W2 + ((size_t)((ntb + ns) * KT + kt) * 32 + lane) * 16;
                v16bf b = *(const v16bf*)bp;
                acc[ns] = __builtin_amdgcn_wmma_f32_16x16x32_bf16(
                    false, a.v, false, b, (short)0, acc[ns], false, false);
            }
        }
    }

    // --- epilogue: residual, relu, store (C/D layout: VGPR r -> row r+8g) ---
#pragma unroll
    for (int ns = 0; ns < NSUB; ++ns) {
#pragma unroll
        for (int r = 0; r < 8; ++r) {
            int row = mt * 16 + r + 8 * g;
            size_t idx = (size_t)row * N + (ntb + ns) * 16 + col;
            float v = acc[ns][r];
            if (resid) v += resid[idx];
            if (relu) v = v > 0.0f ? v : 0.0f;
            if (outF) outF[idx] = v;
            if (outB) outB[idx] = (__bf16)v;
        }
    }
}

// -------------------- CSR-by-destination build (once/launch) ---------------

__global__ __launch_bounds__(256)
void count_kernel(const int* __restrict__ dst, int nE, int dstOff,
                  int* __restrict__ counts) {
    int i = blockIdx.x * 256 + threadIdx.x;
    if (i < nE) atomicAdd(&counts[dst[i] + dstOff], 1);
}

// Single-workgroup chunked exclusive scan: offsets[0..n], offsets[n] = total.
__global__ __launch_bounds__(1024)
void scan_kernel(const int* __restrict__ counts, int* __restrict__ offsets, int n) {
    __shared__ int s[1024];
    int carry = 0;
    for (int base = 0; base < n; base += 1024) {
        int i = base + (int)threadIdx.x;
        int v = (i < n) ? counts[i] : 0;
        s[threadIdx.x] = v;
        __syncthreads();
        for (int d = 1; d < 1024; d <<= 1) {
            int t = (threadIdx.x >= (unsigned)d) ? s[threadIdx.x - d] : 0;
            __syncthreads();
            s[threadIdx.x] += t;
            __syncthreads();
        }
        int incl = s[threadIdx.x];
        if (i < n) offsets[i] = carry + incl - v;
        int total = s[1023];
        __syncthreads();
        carry += total;
    }
    if (threadIdx.x == 0) offsets[n] = carry;
}

__global__ __launch_bounds__(256)
void fill_kernel(const int* __restrict__ src, const int* __restrict__ dst,
                 int nE, int dstOff, int* __restrict__ cursor,
                 int* __restrict__ srcidx) {
    int i = blockIdx.x * 256 + threadIdx.x;
    if (i >= nE) return;
    int pos = atomicAdd(&cursor[dst[i] + dstOff], 1);
    srcidx[pos] = src[i];
}

// ----------------- gather-reduce + mean + bf16 (wave per node) -------------
// Zero fp atomics: each wave streams its CSR row; 32 lanes x float4 = 128
// features, coalesced 512B row reads (h array resident in 192MB L2).
__global__ __launch_bounds__(256)
void gather_mean_kernel(const int* __restrict__ offsets, const int* __restrict__ srcidx,
                        const float* __restrict__ h, __bf16* __restrict__ meanB) {
    int node = blockIdx.x * 8 + ((int)threadIdx.x >> 5);
    if (node >= N_NODES) return;
    int lane = threadIdx.x & 31;
    int beg = offsets[node];
    int end = offsets[node + 1];

    float a0x = 0.f, a0y = 0.f, a0z = 0.f, a0w = 0.f;
    float a1x = 0.f, a1y = 0.f, a1z = 0.f, a1w = 0.f;
    int e = beg;
    for (; e + 4 <= end; e += 4) {
        int s0 = srcidx[e + 0], s1 = srcidx[e + 1];
        int s2 = srcidx[e + 2], s3 = srcidx[e + 3];
        float4 v0 = *(const float4*)(h + (size_t)s0 * EMB + lane * 4);
        float4 v1 = *(const float4*)(h + (size_t)s1 * EMB + lane * 4);
        float4 v2 = *(const float4*)(h + (size_t)s2 * EMB + lane * 4);
        float4 v3 = *(const float4*)(h + (size_t)s3 * EMB + lane * 4);
        a0x += v0.x + v1.x; a0y += v0.y + v1.y;
        a0z += v0.z + v1.z; a0w += v0.w + v1.w;
        a1x += v2.x + v3.x; a1y += v2.y + v3.y;
        a1z += v2.z + v3.z; a1w += v2.w + v3.w;
    }
    for (; e < end; ++e) {
        int s0 = srcidx[e];
        float4 v0 = *(const float4*)(h + (size_t)s0 * EMB + lane * 4);
        a0x += v0.x; a0y += v0.y; a0z += v0.z; a0w += v0.w;
    }
    int cnt = end - beg;
    float inv = (cnt > 0) ? 1.0f / (float)cnt : 0.0f;
    __bf16* o = meanB + (size_t)node * EMB + lane * 4;
    o[0] = (__bf16)((a0x + a1x) * inv);
    o[1] = (__bf16)((a0y + a1y) * inv);
    o[2] = (__bf16)((a0z + a1z) * inv);
    o[3] = (__bf16)((a0w + a1w) * inv);
}

// ------------------------------- launcher ----------------------------------

extern "C" void kernel_launch(void* const* d_in, const int* in_sizes, int n_in,
                              void* d_out, int out_size, void* d_ws, size_t ws_size,
                              hipStream_t stream) {
    (void)in_sizes; (void)n_in; (void)out_size; (void)ws_size;

    // ---- inputs (setup_inputs dict order) ----
    const float* t_ss   = (const float*)d_in[0];
    const float* t_sr   = (const float*)d_in[1];
    const float* t_len  = (const float*)d_in[2];
    const float* t_ct   = (const float*)d_in[3];
    const float* t_mem  = (const float*)d_in[4];
    const float* t_cpu  = (const float*)d_in[5];
    const float* v_ct   = (const float*)d_in[6];
    const float* v_spd  = (const float*)d_in[7];
    const float* v_en   = (const float*)d_in[8];
    const float* v_amem = (const float*)d_in[9];
    const float* v_umf  = (const float*)d_in[10];
    const float* v_acor = (const float*)d_in[11];
    const float* v_ucf  = (const float*)d_in[12];
    const float* v_mem  = (const float*)d_in[13];
    const float* v_cnt  = (const float*)d_in[14];
    const float* v_cor  = (const float*)d_in[15];
    const int*   comp   = (const int*)d_in[16];   // [2, E_TV]  (int32)
    const int*   deps   = (const int*)d_in[17];   // [2, E_DEP] (int32)
    const float* tw1 = (const float*)d_in[18]; const float* tb1 = (const float*)d_in[19];
    const float* tw2 = (const float*)d_in[20]; const float* tb2 = (const float*)d_in[21];
    const float* tw3 = (const float*)d_in[22]; const float* tb3 = (const float*)d_in[23];
    const float* vw1 = (const float*)d_in[24]; const float* vb1 = (const float*)d_in[25];
    const float* vw2 = (const float*)d_in[26]; const float* vb2 = (const float*)d_in[27];
    const float* vw3 = (const float*)d_in[28]; const float* vb3 = (const float*)d_in[29];
    const float* s0_wl = (const float*)d_in[30]; const float* s0_bl = (const float*)d_in[31];
    const float* s0_wr = (const float*)d_in[32];
    const float* s1_wl = (const float*)d_in[33]; const float* s1_bl = (const float*)d_in[34];
    const float* s1_wr = (const float*)d_in[35];

    // ---- workspace arena ----
    char* base = (char*)d_ws;
    size_t off = 0;
    auto alloc = [&](size_t bytes) -> char* {
        char* p = base + off;
        off = (off + bytes + 255) & ~(size_t)255;
        return p;
    };
    float*  maxc    = (float*)alloc(4);
    int*    counts  = (int*)alloc((size_t)N_NODES * 4);
    int*    offsets = (int*)alloc((size_t)(N_NODES + 1) * 4);
    int*    cursor  = (int*)alloc((size_t)N_NODES * 4);
    int*    srcidx  = (int*)alloc((size_t)E_ALL * 4);
    __bf16* taskx   = (__bf16*)alloc((size_t)N_TASKS * 32 * 2);
    __bf16* vmx     = (__bf16*)alloc((size_t)N_VMS * 32 * 2);
    char*   bufA    = alloc((size_t)N_NODES * HID * 2);  // h1 bf16 -> hL1 f32
    char*   bufB    = alloc((size_t)N_NODES * HID * 2);  // h2 bf16 -> mean0B|hL1_bf
    char*   nodeFp  = alloc((size_t)N_NODES * EMB * 4);  // node_x f32
    char*   nodeBp  = alloc((size_t)N_NODES * EMB * 2);  // node_x bf16 -> mean1B
    __bf16* tw1p = (__bf16*)alloc((size_t)32  * HID * 2);
    __bf16* tw2p = (__bf16*)alloc((size_t)HID * HID * 2);
    __bf16* tw3p = (__bf16*)alloc((size_t)HID * EMB * 2);
    __bf16* vw1p = (__bf16*)alloc((size_t)32  * HID * 2);
    __bf16* vw2p = (__bf16*)alloc((size_t)HID * HID * 2);
    __bf16* vw3p = (__bf16*)alloc((size_t)HID * EMB * 2);
    __bf16* wl0p = (__bf16*)alloc((size_t)EMB * EMB * 2);
    __bf16* wr0p = (__bf16*)alloc((size_t)EMB * EMB * 2);
    __bf16* wl1p = (__bf16*)alloc((size_t)EMB * EMB * 2);
    __bf16* wr1p = (__bf16*)alloc((size_t)EMB * EMB * 2);

    // aliased views (non-overlapping lifetimes)
    __bf16* h1_bf  = (__bf16*)bufA;
    __bf16* h2_bf  = (__bf16*)bufB;
    float*  nodeF  = (float*)nodeFp;
    __bf16* nodeB  = (__bf16*)nodeBp;
    __bf16* mean0B = (__bf16*)bufB;                                  // after L3
    __bf16* hL1_bf = (__bf16*)(bufB + (size_t)N_NODES * EMB * 2);
    float*  hL1_f  = (float*)bufA;                                   // after L2
    __bf16* mean1B = (__bf16*)nodeBp;                                // after GEMM0
    float*  outF   = (float*)d_out;

    const int MT_T = N_TASKS / 16;   // 6250
    const int MT_V = N_VMS / 16;     // 125
    const int MT_N = N_NODES / 16;   // 6375

    // ---- 1. max_vm_cores + features ----
    maxc_kernel<<<1, 256, 0, stream>>>(v_cor, N_VMS, maxc);
    task_feat_kernel<<<(N_TASKS + 255) / 256, 256, 0, stream>>>(
        t_ss, t_sr, t_len, t_ct, t_mem, t_cpu, maxc, taskx, N_TASKS);
    vm_feat_kernel<<<(N_VMS + 255) / 256, 256, 0, stream>>>(
        v_ct, v_spd, v_en, v_mem, v_amem, v_umf, v_cnt, v_cor, v_acor, v_ucf,
        maxc, vmx, N_VMS);

    // ---- 2. weight swizzle ----
    auto prep = [&](const float* W, int Korig, int Kpad, int N, __bf16* out) {
        int total = (Kpad >> 5) * (N >> 4) * 32;
        prep_w_kernel<<<(total + 255) / 256, 256, 0, stream>>>(W, Korig, Kpad, N, out);
    };
    prep(tw1, 6,   32,  HID, tw1p);  prep(tw2, HID, HID, HID, tw2p);
    prep(tw3, HID, HID, EMB, tw3p);
    prep(vw1, 10,  32,  HID, vw1p);  prep(vw2, HID, HID, HID, vw2p);
    prep(vw3, HID, HID, EMB, vw3p);
    prep(s0_wl, EMB, EMB, EMB, wl0p); prep(s0_wr, EMB, EMB, EMB, wr0p);
    prep(s1_wl, EMB, EMB, EMB, wl1p); prep(s1_wr, EMB, EMB, EMB, wr1p);

    // ---- 3. CSR-by-destination (built once, reused by both SAGE layers) ----
    zero_int_kernel<<<(N_NODES + 255) / 256, 256, 0, stream>>>(counts, N_NODES);
    count_kernel<<<(E_TV + 255) / 256, 256, 0, stream>>>(comp + E_TV, E_TV, N_TASKS, counts);
    count_kernel<<<(E_DEP + 255) / 256, 256, 0, stream>>>(deps + E_DEP, E_DEP, 0, counts);
    scan_kernel<<<1, 1024, 0, stream>>>(counts, offsets, N_NODES);
    copy_int_kernel<<<(N_NODES + 255) / 256, 256, 0, stream>>>(offsets, cursor, N_NODES);
    fill_kernel<<<(E_TV + 255) / 256, 256, 0, stream>>>(
        comp, comp + E_TV, E_TV, N_TASKS, cursor, srcidx);
    fill_kernel<<<(E_DEP + 255) / 256, 256, 0, stream>>>(
        deps, deps + E_DEP, E_DEP, 0, cursor, srcidx);

    // ---- 4. encoders (WMMA, wave = 16x64 tile) ----
    gemm_wmma_kernel<4><<<dim3(MT_T, HID / 64), 32, 0, stream>>>(
        taskx, tw1p, 32, nullptr, nullptr, 0, tb1, nullptr,
        nullptr, h1_bf, HID, 1);
    gemm_wmma_kernel<4><<<dim3(MT_V, HID / 64), 32, 0, stream>>>(
        vmx, vw1p, 32, nullptr, nullptr, 0, vb1, nullptr,
        nullptr, h1_bf + (size_t)N_TASKS * HID, HID, 1);
    gemm_wmma_kernel<4><<<dim3(MT_T, HID / 64), 32, 0, stream>>>(
        h1_bf, tw2p, HID, nullptr, nullptr, 0, tb2, nullptr,
        nullptr, h2_bf, HID, 1);
    gemm_wmma_kernel<4><<<dim3(MT_V, HID / 64), 32, 0, stream>>>(
        h1_bf + (size_t)N_TASKS * HID, vw2p, HID, nullptr, nullptr, 0, vb2, nullptr,
        nullptr, h2_bf + (size_t)N_TASKS * HID, HID, 1);
    gemm_wmma_kernel<4><<<dim3(MT_T, EMB / 64), 32, 0, stream>>>(
        h2_bf, tw3p, HID, nullptr, nullptr, 0, tb3, nullptr,
        nodeF, nodeB, EMB, 0);
    gemm_wmma_kernel<4><<<dim3(MT_V, EMB / 64), 32, 0, stream>>>(
        h2_bf + (size_t)N_TASKS * HID, vw3p, HID, nullptr, nullptr, 0, vb3, nullptr,
        nodeF + (size_t)N_TASKS * EMB, nodeB + (size_t)N_TASKS * EMB, EMB, 0);

    // ---- 5. SAGE layer 0: gather-mean then fused dual-A WMMA ----
    gather_mean_kernel<<<(N_NODES + 7) / 8, 256, 0, stream>>>(
        offsets, srcidx, nodeF, mean0B);
    gemm_wmma_kernel<4><<<dim3(MT_N, EMB / 64), 32, 0, stream>>>(
        mean0B, wl0p, EMB, nodeB, wr0p, EMB, s0_bl, nodeF,
        hL1_f, hL1_bf, EMB, 1);

    // ---- 6. SAGE layer 1 ----
    gather_mean_kernel<<<(N_NODES + 7) / 8, 256, 0, stream>>>(
        offsets, srcidx, hL1_f, mean1B);
    gemm_wmma_kernel<4><<<dim3(MT_N, EMB / 64), 32, 0, stream>>>(
        mean1B, wl1p, EMB, hL1_bf, wr1p, EMB, s1_bl, hL1_f,
        outF, nullptr, EMB, 1);
}
